// Attn_48137993453608
// MI455X (gfx1250) — matrix operand order repaired and checked by reading, not compiled, not run on hardware
//
#include <hip/hip_runtime.h>

// ---------------------------------------------------------------------------
// Problem constants: B=4, S=2048, D=1024, H=16, DH=64
// ---------------------------------------------------------------------------
#define BB 4
#define SS 2048
#define DD 1024
#define HH 16
#define DHH 64

typedef __attribute__((ext_vector_type(16))) __bf16          v16bf;
typedef __attribute__((ext_vector_type(8)))  float           v8f;
typedef __attribute__((ext_vector_type(8)))  unsigned short  us8;
typedef __attribute__((ext_vector_type(4)))  float           f4;
typedef __attribute__((ext_vector_type(4)))  unsigned short  us4;
typedef __attribute__((ext_vector_type(4)))  int             i32x4;

// ---------------------------------------------------------------------------
// CDNA5 async global->LDS path (probe-confirmed present on this toolchain).
// Builtin wants int32x4 pointers: (int4 as1*, int4 as3*, imm offset, imm cpol)
// ---------------------------------------------------------------------------
#if __has_builtin(__builtin_amdgcn_global_load_async_to_lds_b128)
#define HAVE_ASYNC_LDS 1
#pragma message("CDNA5: global_load_async_to_lds_b128 builtin AVAILABLE - async path enabled")
#else
#define HAVE_ASYNC_LDS 0
#pragma message("CDNA5: global_load_async_to_lds_b128 builtin MISSING - sync fallback")
#endif

#if __has_builtin(__builtin_amdgcn_s_wait_asynccnt)
#define WAIT_ASYNC(n) __builtin_amdgcn_s_wait_asynccnt(n)
#elif HAVE_ASYNC_LDS
#define WAIT_ASYNC(n) asm volatile("s_wait_asynccnt %0" ::"i"(n) : "memory")
#else
#define WAIT_ASYNC(n)
#endif

typedef __attribute__((address_space(1))) i32x4 as1_i32x4;
typedef __attribute__((address_space(3))) i32x4 as3_i32x4;

static __device__ __forceinline__ void async_ld16(const unsigned short* g,
                                                  unsigned short* l) {
#if HAVE_ASYNC_LDS
  __builtin_amdgcn_global_load_async_to_lds_b128(
      (as1_i32x4*)g, (as3_i32x4*)l, 0, 0);
#else
  (void)g; (void)l;
#endif
}

static __device__ __forceinline__ unsigned short f2bf(float f) {
  union { float f; unsigned u; } c; c.f = f;
  unsigned u = c.u;
  u += 0x7FFFu + ((u >> 16) & 1u);   // round-to-nearest-even
  return (unsigned short)(u >> 16);
}

static __device__ __forceinline__ v16bf mk16(us8 lo, us8 hi) {
  union { us8 u[2]; v16bf v; } c;
  c.u[0] = lo; c.u[1] = hi;
  return c.v;
}

// ---------------------------------------------------------------------------
// fp32 -> bf16 bulk convert (vectorized 4-wide)
// ---------------------------------------------------------------------------
__global__ __launch_bounds__(256) void cvt_f32_to_bf16_k(
    const f4* __restrict__ in, us4* __restrict__ out, int n4) {
  int i = blockIdx.x * blockDim.x + threadIdx.x;
  int stride = gridDim.x * blockDim.x;
  for (; i < n4; i += stride) {
    f4 v = in[i];
    us4 r;
    r.x = f2bf(v.x); r.y = f2bf(v.y); r.z = f2bf(v.z); r.w = f2bf(v.w);
    out[i] = r;
  }
}

// ---------------------------------------------------------------------------
// Tiled bf16 WMMA GEMM:  C[m,n] = sum_k A[m,k] * Bw[n,k] + bias[n]
// A: [M,K] bf16 row-major, Bw: [N,K] bf16 row-major (nn.Linear weight layout)
// 256 threads = 8 waves (4x2).  Tile 128x128, K-step 32.
// Async path: double-buffered LDS, 4 async b128s / thread / tile,
//             s_wait_asynccnt 4 overlaps next-tile streaming with WMMA.
// ---------------------------------------------------------------------------
#define TM 128
#define TN 128
#define TK 32
#define LSTR 40   // LDS row stride in halves (80B, 16B aligned, conflict-skewed)

__global__ __launch_bounds__(256) void gemm_bf16_wmma_k(
    const unsigned short* __restrict__ A,
    const unsigned short* __restrict__ Bw,
    const float* __restrict__ bias,
    void* __restrict__ Cout,
    int M, int N, int K, int outIsBf16)
{
#if HAVE_ASYNC_LDS
  __shared__ unsigned short lA[2][TM * LSTR];
  __shared__ unsigned short lB[2][TN * LSTR];
#else
  __shared__ unsigned short lA[1][TM * LSTR];
  __shared__ unsigned short lB[1][TN * LSTR];
#endif

  const int tid   = threadIdx.x;
  const int wave  = tid >> 5;
  const int lane  = tid & 31;
  const int lmod  = lane & 15;
  const int lhalf = lane >> 4;
  const int waveM = wave >> 1;   // 0..3
  const int waveN = wave & 1;    // 0..1
  const int bM = blockIdx.x * TM;
  const int bN = blockIdx.y * TN;

  v8f acc[2][4] = {};

  // cooperative staging: each thread owns half a row (16 halves) of each tile
  const int ldRow = tid >> 1;          // 0..127
  const int ldSeg = (tid & 1) * 16;    // 0 or 16
  const unsigned short* aSrc = A  + (size_t)(bM + ldRow) * K + ldSeg;
  const unsigned short* bSrc = Bw + (size_t)(bN + ldRow) * K + ldSeg;
  const int ldsOff = ldRow * LSTR + ldSeg;

#if HAVE_ASYNC_LDS
  // prologue: stream tile 0 into buffer 0
  async_ld16(aSrc,      &lA[0][ldsOff]);
  async_ld16(aSrc + 8,  &lA[0][ldsOff + 8]);
  async_ld16(bSrc,      &lB[0][ldsOff]);
  async_ld16(bSrc + 8,  &lB[0][ldsOff + 8]);

  int it = 0;
  for (int k0 = 0; k0 < K; k0 += TK, ++it) {
    const int cur = it & 1;
    const bool pre = (k0 + TK) < K;
    if (pre) {                       // stream next tile into other buffer
      const int nb = cur ^ 1;
      const int kn = k0 + TK;
      async_ld16(aSrc + kn,     &lA[nb][ldsOff]);
      async_ld16(aSrc + kn + 8, &lA[nb][ldsOff + 8]);
      async_ld16(bSrc + kn,     &lB[nb][ldsOff]);
      async_ld16(bSrc + kn + 8, &lB[nb][ldsOff + 8]);
      WAIT_ASYNC(4);                 // in-order: current tile's 4 are done
    } else {
      WAIT_ASYNC(0);
    }
    __syncthreads();
#else
  for (int k0 = 0; k0 < K; k0 += TK) {
    const int cur = 0;
    us8 a0 = *(const us8*)(aSrc + k0);
    us8 a1 = *(const us8*)(aSrc + k0 + 8);
    us8 b0 = *(const us8*)(bSrc + k0);
    us8 b1 = *(const us8*)(bSrc + k0 + 8);
    if (k0 + TK < K) {               // hint next K-tile into cache
      __builtin_prefetch(aSrc + k0 + TK, 0, 1);
      __builtin_prefetch(bSrc + k0 + TK, 0, 1);
    }
    *(us8*)(&lA[0][ldsOff])     = a0;
    *(us8*)(&lA[0][ldsOff + 8]) = a1;
    *(us8*)(&lB[0][ldsOff])     = b0;
    *(us8*)(&lB[0][ldsOff + 8]) = b1;
    __syncthreads();
#endif

    // A fragments: lane holds row M=lmod; K = lhalf*16 + [0..15] contiguous
    v16bf af[2];
#pragma unroll
    for (int i = 0; i < 2; ++i) {
      int row = waveM * 32 + i * 16 + lmod;
      int kb = lhalf * 16;
      us8 lo = *(const us8*)(&lA[cur][row * LSTR + kb]);
      us8 hi = *(const us8*)(&lA[cur][row * LSTR + kb + 8]);
      af[i] = mk16(lo, hi);
    }
    // B fragments: lane holds col N=lmod; K = lhalf*16 + [0..15] contiguous
#pragma unroll
    for (int j = 0; j < 4; ++j) {
      int col = waveN * 64 + j * 16 + lmod;
      int kb = lhalf * 16;
      us8 lo = *(const us8*)(&lB[cur][col * LSTR + kb]);
      us8 hi = *(const us8*)(&lB[cur][col * LSTR + kb + 8]);
      v16bf bf = mk16(lo, hi);
#pragma unroll
      for (int i = 0; i < 2; ++i) {
        acc[i][j] = __builtin_amdgcn_wmma_f32_16x16x32_bf16(
            false, af[i], false, bf, (short)0, acc[i][j], false, false);
      }
    }
    __syncthreads();   // all waves done reading 'cur' before it is refilled
  }

  // epilogue: C layout -> M = r + 8*lhalf (+frag base), N = lmod (+frag base)
#pragma unroll
  for (int j = 0; j < 4; ++j) {
    const int n = bN + waveN * 64 + j * 16 + lmod;
    const float bv = bias[n];
#pragma unroll
    for (int i = 0; i < 2; ++i) {
      const int m0 = bM + waveM * 32 + i * 16 + lhalf * 8;
#pragma unroll
      for (int r = 0; r < 8; ++r) {
        float v = acc[i][j][r] + bv;
        if (outIsBf16) ((unsigned short*)Cout)[(size_t)(m0 + r) * N + n] = f2bf(v);
        else           ((float*)Cout)[(size_t)(m0 + r) * N + n] = v;
      }
    }
  }
}

// ---------------------------------------------------------------------------
// Stage 2: per (b,h)  w[d,e] = softmax_e( sum_s Q[s,d]*K[s,e] / sqrt(S) )
// Q/K columns of Y (bf16 [B*S, 3D]).  1 block per (b,h), 4 waves.
// LDS tiles are stored transposed ([col][s]) so WMMA fragments read
// contiguous K(=s) runs.  K-loop: S=2048 in steps of 32.
// ---------------------------------------------------------------------------
__global__ __launch_bounds__(128) void attn_qk_softmax_k(
    const unsigned short* __restrict__ Y,   // [B*S, 3D] bf16
    unsigned short* __restrict__ Wout)      // [B*H, 64, 64] bf16
{
  __shared__ unsigned short Ql[DHH * LSTR];
  __shared__ unsigned short Kl[DHH * LSTR];

  const int bh = blockIdx.x;
  const int b = bh >> 4, h = bh & 15;
  const int qc = h * DHH;
  const int kc = DD + h * DHH;
  const int tid = threadIdx.x;
  const int wave = tid >> 5, lane = tid & 31;
  const int lmod = lane & 15, lhalf = lane >> 4;

  v8f acc[4] = {};

  for (int s0 = 0; s0 < SS; s0 += 32) {
    // load 32(s) x 64(col) tiles of Q and K, transposed into LDS [col][s]
#pragma unroll
    for (int itr = 0; itr < 2; ++itr) {
      int idx = tid * 2 + itr;         // 0..255
      int s  = idx >> 3;               // 0..31
      int c8 = (idx & 7) * 8;          // 0..56
      const size_t rowOff = (size_t)(b * SS + s0 + s) * (3 * DD);
      us8 qv = *(const us8*)(Y + rowOff + qc + c8);
      us8 kv = *(const us8*)(Y + rowOff + kc + c8);
#pragma unroll
      for (int e = 0; e < 8; ++e) {
        Ql[(c8 + e) * LSTR + s] = qv[e];
        Kl[(c8 + e) * LSTR + s] = kv[e];
      }
    }
    __syncthreads();

    // A = Q^T (d x s), wave w owns rows d in [16w, 16w+16)
    v16bf af;
    {
      int d = wave * 16 + lmod;
      int kb = lhalf * 16;
      us8 lo = *(const us8*)(&Ql[d * LSTR + kb]);
      us8 hi = *(const us8*)(&Ql[d * LSTR + kb + 8]);
      af = mk16(lo, hi);
    }
#pragma unroll
    for (int j = 0; j < 4; ++j) {
      int e = j * 16 + lmod;
      int kb = lhalf * 16;
      us8 lo = *(const us8*)(&Kl[e * LSTR + kb]);
      us8 hi = *(const us8*)(&Kl[e * LSTR + kb + 8]);
      v16bf bf = mk16(lo, hi);
      acc[j] = __builtin_amdgcn_wmma_f32_16x16x32_bf16(
          false, af, false, bf, (short)0, acc[j], false, false);
    }
    __syncthreads();
  }

  // scale + softmax: row M lives across the 16 lanes of one half-wave
  const float scale = 0.022097086912079608f;   // 1/sqrt(2048)
#pragma unroll
  for (int r = 0; r < 8; ++r) {
    float ex[4];
    float mx = -3.4e38f;
#pragma unroll
    for (int j = 0; j < 4; ++j) { ex[j] = acc[j][r] * scale; mx = fmaxf(mx, ex[j]); }
#pragma unroll
    for (int dxor = 8; dxor >= 1; dxor >>= 1) mx = fmaxf(mx, __shfl_xor(mx, dxor, 32));
    float sm = 0.f;
#pragma unroll
    for (int j = 0; j < 4; ++j) { ex[j] = __expf(ex[j] - mx); sm += ex[j]; }
#pragma unroll
    for (int dxor = 8; dxor >= 1; dxor >>= 1) sm += __shfl_xor(sm, dxor, 32);
    float inv = 1.0f / sm;
    int m = wave * 16 + r + lhalf * 8;
    unsigned short* dst = Wout + (size_t)bh * (DHH * DHH) + m * DHH;
#pragma unroll
    for (int j = 0; j < 4; ++j) dst[j * 16 + lmod] = f2bf(ex[j] * inv);
  }
}

// ---------------------------------------------------------------------------
// Stage 3: per (b,h)  out[d,s] = sum_e w[d,e] * V[s,e]   (M=64, N=2048, K=64)
// B-operand (V) reads K(=e) contiguous straight from Y -> no transpose.
// Epilogue implements the raw [B,H,DH,S] -> [B,S,D] reshape via index remap.
// Grid: (B*H, S/128), 8 waves each owning one 16-wide s fragment.
// ---------------------------------------------------------------------------
#define WSTR 72   // LDS stride for the 64x64 w tile

__global__ __launch_bounds__(256) void attn_ov_k(
    const unsigned short* __restrict__ Y,      // [B*S, 3D] bf16
    const unsigned short* __restrict__ Wsoft,  // [B*H, 64, 64] bf16
    unsigned short* __restrict__ O)            // [B*S, D] bf16 (remapped)
{
  __shared__ unsigned short wl[DHH * WSTR];

  const int bh = blockIdx.x;
  const int b = bh >> 4, h = bh & 15;
  const int vc = 2 * DD + h * DHH;
  const int tid = threadIdx.x;
  const int wave = tid >> 5, lane = tid & 31;
  const int lmod = lane & 15, lhalf = lane >> 4;

  // stage w tile (4096 halves) into LDS
#pragma unroll
  for (int itr = 0; itr < 2; ++itr) {
    int idx = tid + itr * 256;       // 0..511
    int d  = idx >> 3;               // 0..63
    int c8 = (idx & 7) * 8;          // 0..56
    const unsigned short* src = Wsoft + (size_t)bh * (DHH * DHH) + d * DHH + c8;
#if HAVE_ASYNC_LDS
    async_ld16(src, &wl[d * WSTR + c8]);
#else
    *(us8*)(&wl[d * WSTR + c8]) = *(const us8*)src;
#endif
  }
#if HAVE_ASYNC_LDS
  WAIT_ASYNC(0);
#endif
  __syncthreads();

  const int sFrag = blockIdx.y * 128 + wave * 16;
  v8f acc[4] = {};

#pragma unroll
  for (int e0 = 0; e0 < DHH; e0 += 32) {
    // B fragment: lane = s column, K = e contiguous from global
    v16bf bf;
    {
      int s = sFrag + lmod;
      int ke = e0 + lhalf * 16;
      const unsigned short* src = Y + (size_t)(b * SS + s) * (3 * DD) + vc + ke;
      us8 lo = *(const us8*)(src);
      us8 hi = *(const us8*)(src + 8);
      bf = mk16(lo, hi);
    }
#pragma unroll
    for (int i = 0; i < 4; ++i) {
      int d = i * 16 + lmod;
      int kb = e0 + lhalf * 16;
      us8 lo = *(const us8*)(&wl[d * WSTR + kb]);
      us8 hi = *(const us8*)(&wl[d * WSTR + kb + 8]);
      v16bf af = mk16(lo, hi);
      acc[i] = __builtin_amdgcn_wmma_f32_16x16x32_bf16(
          false, af, false, bf, (short)0, acc[i], false, false);
    }
  }

  // epilogue with raw reshape: flat = (h*64+d)*S + s ; (s2,c2) = flat /% D
  const int s = sFrag + lmod;
#pragma unroll
  for (int i = 0; i < 4; ++i) {
#pragma unroll
    for (int r = 0; r < 8; ++r) {
      int dl = i * 16 + r + lhalf * 8;
      unsigned F = (unsigned)((h * DHH + dl) * SS + s);
      unsigned s2 = F >> 10;          // / D
      unsigned c2 = F & 1023u;        // % D
      O[((size_t)b * SS + s2) * DD + c2] = f2bf(acc[i][r]);
    }
  }
}

// ---------------------------------------------------------------------------
// Host-side launch
// ---------------------------------------------------------------------------
extern "C" void kernel_launch(void* const* d_in, const int* in_sizes, int n_in,
                              void* d_out, int out_size, void* d_ws, size_t ws_size,
                              hipStream_t stream) {
  (void)in_sizes; (void)n_in; (void)out_size; (void)ws_size;

  const float* X  = (const float*)d_in[0];   // [B,S,D]
  const float* W1 = (const float*)d_in[1];   // [3D,D]
  const float* b1 = (const float*)d_in[2];   // [3D]
  const float* W2 = (const float*)d_in[3];   // [D,D]
  const float* b2 = (const float*)d_in[4];   // [D]

  const size_t nX  = (size_t)BB * SS * DD;        // 8388608
  const size_t nW1 = (size_t)3 * DD * DD;         // 3145728
  const size_t nW2 = (size_t)DD * DD;             // 1048576
  const size_t nY  = (size_t)BB * SS * 3 * DD;    // 25165824
  const size_t nWs = (size_t)BB * HH * DHH * DHH; // 262144

  char* w = (char*)d_ws;
  unsigned short* Xb  = (unsigned short*)w;                    w += nX  * 2;
  unsigned short* W1b = (unsigned short*)w;                    w += nW1 * 2;
  unsigned short* W2b = (unsigned short*)w;                    w += nW2 * 2;
  unsigned short* Yb  = (unsigned short*)w;                    w += nY  * 2;
  unsigned short* Wsm = (unsigned short*)w;                    w += nWs * 2;
  unsigned short* Ob  = (unsigned short*)w;                    /* nX*2 */

  // fp32 -> bf16 converts
  cvt_f32_to_bf16_k<<<dim3(2048), 256, 0, stream>>>((const f4*)X,  (us4*)Xb,  (int)(nX  / 4));
  cvt_f32_to_bf16_k<<<dim3(1024), 256, 0, stream>>>((const f4*)W1, (us4*)W1b, (int)(nW1 / 4));
  cvt_f32_to_bf16_k<<<dim3(512),  256, 0, stream>>>((const f4*)W2, (us4*)W2b, (int)(nW2 / 4));

  // GEMM1: Y = X @ W1^T + b1   (M=8192, N=3072, K=1024) -> bf16
  gemm_bf16_wmma_k<<<dim3(8192 / TM, 3072 / TN), 256, 0, stream>>>(
      Xb, W1b, b1, (void*)Yb, BB * SS, 3 * DD, DD, 1);

  // Attention: w = softmax(q^T k / sqrt(S)) per (b,h)
  attn_qk_softmax_k<<<dim3(BB * HH), 128, 0, stream>>>(Yb, Wsm);

  // out = w @ v  with fused raw reshape -> Ob [B,S,D] bf16
  attn_ov_k<<<dim3(BB * HH, SS / 128), 256, 0, stream>>>(Yb, Wsm, Ob);

  // GEMM2: final = Ob @ W2^T + b2  (M=8192, N=1024, K=1024) -> fp32 d_out
  gemm_bf16_wmma_k<<<dim3(8192 / TM, 1024 / TN), 256, 0, stream>>>(
      Ob, W2b, b2, d_out, BB * SS, DD, DD, 0);
}
